// LSTMClassifier_68204080660801
// MI455X (gfx1250) — 16-bit and 8-bit matrix operand order repaired and checked by reading, compiled, not run on hardware
//
#include <hip/hip_runtime.h>
#include <hip/hip_bf16.h>


typedef __attribute__((ext_vector_type(16))) __bf16 v16bf;
typedef __attribute__((ext_vector_type(8)))  __bf16 v8bf;
typedef __attribute__((ext_vector_type(8)))  float  v8f;
typedef __attribute__((ext_vector_type(16))) int    v16i;

#define T_STEPS 2048
#define BATCH   256
#define DIM     128
#define HID     256
#define NT      64

#define WS_WHH  0
#define WS_WIH  (262144)
#define WS_BIAS (262144 + 262144)
#define WS_XP   (262144 + 262144 + 4096)

#define LDS_WHH_BYTES (64 * 2 * 2048)
#define LDS_H_OFF     LDS_WHH_BYTES
#define LDS_H_BYTES   4096
#define LDS_OUT_OFF   (LDS_H_OFF + LDS_H_BYTES)
#define LDS_TOTAL     (LDS_OUT_OFF + 16 * 2 * 4)

__device__ inline unsigned char f32_e4m3(float x) {
  union { float f; unsigned u; } v; v.f = x;
  unsigned s = (v.u >> 24) & 0x80u;
  float a = __builtin_fabsf(x);
  if (a < 0.015625f) {
    int q = (int)(a * 512.0f + 0.5f);
    return (unsigned char)(s | (unsigned)q);
  }
  v.f = a;
  int e = (int)((v.u >> 23) & 0xFFu) - 127;
  unsigned man = v.u & 0x7FFFFFu;
  unsigned m3  = man >> 20;
  unsigned rem = man & 0xFFFFFu;
  unsigned eb  = (unsigned)(e + 7);
  if (rem > 0x80000u || (rem == 0x80000u && (m3 & 1u))) { m3++; if (m3 == 8u) { m3 = 0u; eb++; } }
  if (eb > 15u || (eb == 15u && m3 > 6u)) return (unsigned char)(s | 0x7Eu);
  return (unsigned char)(s | (eb << 3) | m3);
}

__device__ inline float fsig(float x)  { return 1.0f / (1.0f + __expf(-x)); }
__device__ inline float ftanhf(float x){ return 2.0f / (1.0f + __expf(-2.0f * x)) - 1.0f; }

__global__ void k_prep_wih(const float* __restrict__ Wih, __bf16* __restrict__ out) {
  int idx  = blockIdx.x * blockDim.x + threadIdx.x;
  int lane = idx & 31;
  int kt   = (idx >> 5) & 3;
  int nt   = idx >> 7;
  int n  = nt * 16 + (lane & 15);
  int kb = kt * 32 + ((lane >> 4) << 4);
  v16bf o;
#pragma unroll
  for (int e = 0; e < 16; ++e) o[e] = (__bf16)Wih[n * DIM + kb + e];
  *(v16bf*)(out + (size_t)idx * 16) = o;
}

__global__ void k_prep_whh(const float* __restrict__ Whh, unsigned char* __restrict__ out) {
  int idx  = blockIdx.x * blockDim.x + threadIdx.x;
  int lane = idx & 31;
  int kt   = (idx >> 5) & 1;
  int nt   = idx >> 6;
  int n    = nt * 16 + (lane & 15);
  int kadd = ((lane >> 4) << 4) + kt * 128;
  unsigned wds[16];
#pragma unroll
  for (int d = 0; d < 16; ++d) {
    unsigned wrd = 0;
#pragma unroll
    for (int b = 0; b < 4; ++b) {
      int j = d * 4 + b;
      int v = j >> 2;
      int k = ((v & 3) << 2) + (j & 3) + ((v >> 2) << 5) + kadd;
      wrd |= ((unsigned)f32_e4m3(Whh[n * HID + k])) << (8 * b);
    }
    wds[d] = wrd;
  }
  unsigned* o = (unsigned*)(out + (size_t)idx * 64);
#pragma unroll
  for (int d = 0; d < 16; ++d) o[d] = wds[d];
}

__global__ void k_prep_bias(const float* __restrict__ bih, const float* __restrict__ bhh,
                            float* __restrict__ out) {
  int i = blockIdx.x * blockDim.x + threadIdx.x;
  if (i < 1024) out[i] = bih[i] + bhh[i];
}

__global__ void k_xproj(const float* __restrict__ x, const __bf16* __restrict__ WihSw,
                        const float* __restrict__ biasC, __bf16* __restrict__ xp) {
  const int mt   = blockIdx.x;
  const int w    = threadIdx.x >> 5;
  const int lane = threadIdx.x & 31;
  const int hf   = lane >> 4;
  const int ncol = lane & 15;
  const size_t m = (size_t)mt * 16 + ncol;

  v16bf a[4];
#pragma unroll
  for (int kt = 0; kt < 4; ++kt) {
    const float* ap = x + m * DIM + kt * 32 + hf * 16;
    float4 c0 = *(const float4*)(ap + 0);
    float4 c1 = *(const float4*)(ap + 4);
    float4 c2 = *(const float4*)(ap + 8);
    float4 c3 = *(const float4*)(ap + 12);
    a[kt][0]=(__bf16)c0.x;  a[kt][1]=(__bf16)c0.y;  a[kt][2]=(__bf16)c0.z;  a[kt][3]=(__bf16)c0.w;
    a[kt][4]=(__bf16)c1.x;  a[kt][5]=(__bf16)c1.y;  a[kt][6]=(__bf16)c1.z;  a[kt][7]=(__bf16)c1.w;
    a[kt][8]=(__bf16)c2.x;  a[kt][9]=(__bf16)c2.y;  a[kt][10]=(__bf16)c2.z; a[kt][11]=(__bf16)c2.w;
    a[kt][12]=(__bf16)c3.x; a[kt][13]=(__bf16)c3.y; a[kt][14]=(__bf16)c3.z; a[kt][15]=(__bf16)c3.w;
  }

#pragma unroll
  for (int j = 0; j < 16; ++j) {
    const int nt = w * 16 + j;
    v8f acc = {};
#pragma unroll
    for (int kt = 0; kt < 4; ++kt) {
      v16bf b = *(const v16bf*)(WihSw + ((size_t)(nt * 4 + kt) * 32 + lane) * 16);
      acc = __builtin_amdgcn_wmma_f32_16x16x32_bf16(false, a[kt], false, b,
                                                    (short)0, acc, false, false);
    }
    float bias = biasC[nt * 16 + ncol];
    v8bf o;
#pragma unroll
    for (int r = 0; r < 8; ++r) o[r] = (__bf16)(acc[r] + bias);
    *(v8bf*)(xp + (((size_t)mt * NT + nt) * 32 + lane) * 8) = o;
  }
}

__global__ void k_lstm(const unsigned char* __restrict__ WhhSw,
                       const __bf16* __restrict__ xp,
                       const float* __restrict__ Wout,
                       const float* __restrict__ bout,
                       float* __restrict__ out) {
  extern __shared__ unsigned char smem[];
  unsigned char* sW  = smem;
  unsigned char* sH  = smem + LDS_H_OFF;
  float*         sOut = (float*)(smem + LDS_OUT_OFF);

  const int bt   = blockIdx.x;
  const int tid  = threadIdx.x;
  const int w    = tid >> 5;
  const int lane = tid & 31;
  const int hfD  = lane >> 4;
  const int nlo  = lane & 15;

  {
    const uint4* src = (const uint4*)WhhSw;
    uint4*       dst = (uint4*)sW;
    for (int i = tid; i < LDS_WHH_BYTES / 16; i += 128) dst[i] = src[i];
  }
  for (int i = tid; i < LDS_H_BYTES / 4; i += 128) ((unsigned*)sH)[i] = 0u;
  if (tid < 32) sOut[tid] = 0.0f;
  __syncthreads();

  v8f creg[4] = {};
  v8f hreg[4] = {};

  for (int t = 0; t < T_STEPS; ++t) {
    v16i a0 = *(const v16i*)(sH + lane * 64);
    v16i a1 = *(const v16i*)(sH + 2048 + lane * 64);
    __syncthreads();

    const __bf16* xpt = xp + ((size_t)t * 16 + bt) * (NT * 256);
    if (t + 1 < T_STEPS) {
      const unsigned char* pf =
          (const unsigned char*)(xp + ((size_t)(t + 1) * 16 + bt) * (NT * 256)) + tid * 64;
      __builtin_prefetch(pf, 0, 0);
      __builtin_prefetch(pf + 8192, 0, 0);
      __builtin_prefetch(pf + 16384, 0, 0);
      __builtin_prefetch(pf + 24576, 0, 0);
    }

#pragma unroll
    for (int q = 0; q < 4; ++q) {
      v8f acc[4];
#pragma unroll
      for (int g = 0; g < 4; ++g) {
        const int nt = g * 16 + w * 4 + q;
        v8bf xv = *(const v8bf*)(xpt + ((size_t)nt * 32 + lane) * 8);
        v8f a;
#pragma unroll
        for (int r = 0; r < 8; ++r) a[r] = (float)xv[r];
        v16i b0 = *(const v16i*)(sW + (size_t)(nt * 2 + 0) * 2048 + lane * 64);
        a = __builtin_amdgcn_wmma_f32_16x16x128_fp8_fp8(a0, b0, (short)0, a, false, false);
        v16i b1 = *(const v16i*)(sW + (size_t)(nt * 2 + 1) * 2048 + lane * 64);
        a = __builtin_amdgcn_wmma_f32_16x16x128_fp8_fp8(a1, b1, (short)0, a, false, false);
        acc[g] = a;
      }
      v8f hv;
#pragma unroll
      for (int r = 0; r < 8; ++r) {
        float iv = fsig(acc[0][r]);
        float fv = fsig(acc[1][r]);
        float gv = ftanhf(acc[2][r]);
        float ov = fsig(acc[3][r]);
        float cn = fv * creg[q][r] + iv * gv;
        creg[q][r] = cn;
        hv[r] = ov * ftanhf(cn);
      }
      hreg[q] = hv;
      const int k    = w * 64 + q * 16 + nlo;
      const int kt   = k >> 7;
      const int k64  = k & 63;
      const int lh   = (k64 >> 4) & 1;
      const int kp   = k64 - (lh << 4);
      const int e    = ((kp >> 5) << 4) + (kp & 15);
      const int base = kt * 2048 + (((k & 127) >> 6) << 5) + e;
#pragma unroll
      for (int r = 0; r < 8; ++r) {
        int laneA = (r + 8 * hfD) + 16 * lh;
        sH[base + laneA * 64] = f32_e4m3(hv[r]);
      }
    }
    __syncthreads();
  }

  float p0[8] = {}, p1[8] = {};
#pragma unroll
  for (int q = 0; q < 4; ++q) {
    int j = w * 64 + q * 16 + nlo;
    float w0 = Wout[j];
    float w1 = Wout[256 + j];
#pragma unroll
    for (int r = 0; r < 8; ++r) {
      p0[r] += hreg[q][r] * w0;
      p1[r] += hreg[q][r] * w1;
    }
  }
#pragma unroll
  for (int r = 0; r < 8; ++r) {
    int m = r + 8 * hfD;
    atomicAdd(&sOut[m * 2 + 0], p0[r]);
    atomicAdd(&sOut[m * 2 + 1], p1[r]);
  }
  __syncthreads();
  if (tid < 32) {
    int m = tid >> 1, l = tid & 1;
    float v = sOut[tid] + bout[l];
    out[(size_t)(bt * 16 + m) * 2 + l] = fsig(v);
  }
}

extern "C" void kernel_launch(void* const* d_in, const int* in_sizes, int n_in,
                              void* d_out, int out_size, void* d_ws, size_t ws_size,
                              hipStream_t stream) {
  (void)in_sizes; (void)n_in; (void)out_size; (void)ws_size;
  const float* x    = (const float*)d_in[0];
  const float* Wih  = (const float*)d_in[1];
  const float* Whh  = (const float*)d_in[2];
  const float* bih  = (const float*)d_in[3];
  const float* bhh  = (const float*)d_in[4];
  const float* Wout = (const float*)d_in[5];
  const float* bo   = (const float*)d_in[6];
  float* out = (float*)d_out;

  unsigned char* ws = (unsigned char*)d_ws;
  unsigned char* wsWhh  = ws + WS_WHH;
  __bf16*        wsWih  = (__bf16*)(ws + WS_WIH);
  float*         wsBias = (float*)(ws + WS_BIAS);
  __bf16*        wsXp   = (__bf16*)(ws + WS_XP);

  hipFuncSetAttribute((const void*)k_lstm,
                      hipFuncAttributeMaxDynamicSharedMemorySize, LDS_TOTAL);

  k_prep_whh <<<dim3(16),    256, 0, stream>>>(Whh, wsWhh);
  k_prep_wih <<<dim3(32),    256, 0, stream>>>(Wih, wsWih);
  k_prep_bias<<<dim3(4),     256, 0, stream>>>(bih, bhh, wsBias);
  k_xproj    <<<dim3(32768), 128, 0, stream>>>(x, wsWih, wsBias, wsXp);
  k_lstm     <<<dim3(16),    128, LDS_TOTAL, stream>>>(wsWhh, wsXp, Wout, bo, out);
}
